// GraphConvBottleneck_37538014167435
// MI455X (gfx1250) — matrix operand order repaired and checked by reading, not compiled, not run on hardware
//
#include <hip/hip_runtime.h>

// ---------------------------------------------------------------------------
// GraphConvBottleneck on MI455X (gfx1250, wave32, WMMA)
//
// y = adj @ (x W) + x Lw + b   per layer; adj is 8192x8192 dense.
//   * adj converted once to bf16 (128 MB -> L2-resident, reused by 6 layers)
//   * per layer: projection kernel -> XW^T (bf16,[B,64,N]) + XLW^T (f32)
//   * WMMA GEMM: wave tile = 32 rows x 64 cols (2x4 16x16 accumulators),
//     software-pipelined (double-buffered fragments) so loads of step k+1
//     overlap the v_wmma_f32_16x16x32_bf16 of step k.
// ---------------------------------------------------------------------------

typedef __attribute__((ext_vector_type(16))) __bf16        v16bf;
typedef __attribute__((ext_vector_type(8)))  float         v8f;
typedef __attribute__((ext_vector_type(4)))  unsigned int  uint4v;

#define GN 8192
#define GB 16
#define GH 64

union BfFrag {
    uint4v u[2];
    v16bf  v;
};

__device__ __forceinline__ unsigned short f2bf(float f) {
    unsigned int u = __float_as_uint(f);
    unsigned int r = u + 0x7FFFu + ((u >> 16) & 1u);  // round-to-nearest-even
    return (unsigned short)(r >> 16);
}

__device__ __forceinline__ v8f wmma_bf16(const v16bf& a, const v16bf& b, v8f c) {
    return __builtin_amdgcn_wmma_f32_16x16x32_bf16(false, a, false, b,
                                                   (short)0, c, false, false);
}

// ---------------------------------------------------------------------------
// adj f32 -> bf16, 4 elements per thread
// ---------------------------------------------------------------------------
__global__ __launch_bounds__(256) void cvt_adj_bf16(
        const float* __restrict__ a, unsigned short* __restrict__ o, long n) {
    long i = ((long)blockIdx.x * blockDim.x + threadIdx.x) * 4L;
    if (i + 3 < n) {
        float4 f = *(const float4*)(a + i);
        ushort4 r;
        r.x = f2bf(f.x); r.y = f2bf(f.y); r.z = f2bf(f.z); r.w = f2bf(f.w);
        *(ushort4*)(o + i) = r;
    }
}

// ---------------------------------------------------------------------------
// Projection: XW^T[b][h][n] = bf16(sum_k x[b][n][k] w[k][h]) (0 for h>=Dout)
//             XLW^T[b][h][n] = sum_k x[b][n][k] lw[k][h]
// 256 threads: 64 rows (n) x 64 cols (h); x tile staged in LDS (stride 65).
// ---------------------------------------------------------------------------
__global__ __launch_bounds__(256) void gcn_proj(
        const float* __restrict__ x, const float* __restrict__ w,
        const float* __restrict__ lw, unsigned short* __restrict__ ytT,
        float* __restrict__ xlwT, int Din, int Dout) {
    __shared__ float xs[64 * 65];
    const int b  = blockIdx.y;
    const int n0 = blockIdx.x * 64;
    const int t  = threadIdx.x;

    const int total = 64 * Din;
    for (int i = t; i < total; i += 256) {
        int r = i / Din;
        int k = i - r * Din;
        xs[r * 65 + k] = x[((size_t)b * GN + (n0 + r)) * (size_t)Din + k];
    }
    __syncthreads();

    const int nl = t & 63;   // contiguous across lanes -> coalesced stores
    const int hb = t >> 6;   // 0..3, wave-uniform

    for (int j = 0; j < 16; ++j) {
        int h = hb + 4 * j;  // wave-uniform column -> scalar weight loads
        float sw = 0.f, sl = 0.f;
        if (h < Dout) {
            for (int k = 0; k < Din; ++k) {
                float xv = xs[nl * 65 + k];
                sw = fmaf(xv, w[k * Dout + h], sw);
                sl = fmaf(xv, lw[k * Dout + h], sl);
            }
        }
        size_t ti = ((size_t)b * GH + h) * GN + n0 + nl;
        ytT[ti]  = f2bf(sw);
        xlwT[ti] = sl;
    }
}

// ---------------------------------------------------------------------------
// Main GEMM: Z[b] = adj(bf16) @ XW[b].
// Block = 8 waves, 256 rows. Wave tile = 2 row-tiles x CT col-tiles.
// Double-buffered fragment registers -> loads overlap WMMAs.
// Fused epilogue: z += XLW + bias; optional relu; optional (res+z)*0.5.
// ---------------------------------------------------------------------------
template <int CT>
__global__ __launch_bounds__(256) void gcn_wmma(
        const unsigned short* __restrict__ adjb,
        const unsigned short* __restrict__ ytT,
        const float* __restrict__ xlwT,
        const float* __restrict__ bias,
        const float* __restrict__ resid,
        float* __restrict__ out,
        int outDim, int doRelu, int doResid) {
    const int wave = threadIdx.x >> 5;
    const int lane = threadIdx.x & 31;
    const int half = lane >> 4;      // 0: lanes 0-15, 1: lanes 16-31
    const int l16  = lane & 15;
    const int b    = blockIdx.y;
    const int rowBase = blockIdx.x * 256 + wave * 32;

    // A fragments: adj rows (row-major bf16); M = l16 in both halves.
    const unsigned short* aPtr0 = adjb + (size_t)(rowBase + l16) * GN;
    const unsigned short* aPtr1 = adjb + (size_t)(rowBase + 16 + l16) * GN;
    const unsigned short* bBase = ytT + (size_t)b * GH * GN;

    v8f acc[2][CT];
#pragma unroll
    for (int rt = 0; rt < 2; ++rt)
#pragma unroll
        for (int t = 0; t < CT; ++t)
#pragma unroll
            for (int i = 0; i < 8; ++i) acc[rt][t][i] = 0.f;

    // A 16x32 bf16 layout: half 0 -> K 0..7 | 16..23 ; half 1 -> K 8..15 | 24..31
    auto loadA = [&](BfFrag* A, int k0) {
        A[0].u[0] = *(const uint4v*)(aPtr0 + k0 + half * 8);
        A[0].u[1] = *(const uint4v*)(aPtr0 + k0 + 16 + half * 8);
        A[1].u[0] = *(const uint4v*)(aPtr1 + k0 + half * 8);
        A[1].u[1] = *(const uint4v*)(aPtr1 + k0 + 16 + half * 8);
    };
    // B 32x16 bf16: col = l16; half 0 -> K 0..7 | 16..23 ; half 1 -> K 8..15 | 24..31
    // (same order as A).
    auto loadB = [&](BfFrag* Bf, int k0) {
#pragma unroll
        for (int t = 0; t < CT; ++t) {
            const unsigned short* bp =
                bBase + (size_t)(t * 16 + l16) * GN + k0 + half * 8;
            Bf[t].u[0] = ((const uint4v*)bp)[0];
            Bf[t].u[1] = ((const uint4v*)bp)[2];
        }
    };
    auto mm = [&](BfFrag* A, BfFrag* Bf) {
#pragma unroll
        for (int t = 0; t < CT; ++t) {
            acc[0][t] = wmma_bf16(A[0].v, Bf[t].v, acc[0][t]);
            acc[1][t] = wmma_bf16(A[1].v, Bf[t].v, acc[1][t]);
        }
    };

    BfFrag a0[2], a1[2], b0[CT], b1[CT];
    loadA(a0, 0);
    loadB(b0, 0);
    for (int k0 = 0; k0 < GN; k0 += 64) {
        __builtin_prefetch(aPtr0 + k0 + 512, 0, 3);
        __builtin_prefetch(aPtr1 + k0 + 512, 0, 3);
        loadA(a1, k0 + 32);          // stage k0+32 while computing k0
        loadB(b1, k0 + 32);
        mm(a0, b0);
        if (k0 + 64 < GN) {          // stage k0+64 while computing k0+32
            loadA(a0, k0 + 64);
            loadB(b0, k0 + 64);
        }
        mm(a1, b1);
    }

    // Epilogue. C layout: VGPR r -> row (half ? 8 : 0) + r, col = l16.
#pragma unroll
    for (int rt = 0; rt < 2; ++rt) {
#pragma unroll
        for (int t = 0; t < CT; ++t) {
            const int col = t * 16 + l16;
            if (col >= outDim) continue;
            const float bs = bias[col];
            const float* xl = xlwT + ((size_t)b * GH + col) * GN;
#pragma unroll
            for (int r = 0; r < 8; ++r) {
                const int row = rowBase + rt * 16 + half * 8 + r;
                float v = acc[rt][t][r] + xl[row] + bs;
                if (doRelu) v = fmaxf(v, 0.f);
                if (doResid)
                    v = 0.5f * (resid[((size_t)b * GN + row) * GH + col] + v);
                out[((size_t)b * GN + row) * (size_t)outDim + col] = v;
            }
        }
    }
}

__global__ __launch_bounds__(256) void copy_f32(
        const float* __restrict__ in, float* __restrict__ out, long n) {
    long i = (long)blockIdx.x * blockDim.x + threadIdx.x;
    if (i < n) out[i] = in[i];
}

// ---------------------------------------------------------------------------
extern "C" void kernel_launch(void* const* d_in, const int* in_sizes, int n_in,
                              void* d_out, int out_size, void* d_ws, size_t ws_size,
                              hipStream_t stream) {
    const int B = GB, N = GN, IN = 3, H = GH, OUT = 3, NB = 2;

    const float* verts  = (const float*)d_in[0];
    const float* adj    = (const float*)d_in[1];
    const float* w_in   = (const float*)d_in[2];
    const float* lw_in  = (const float*)d_in[3];
    const float* b_in   = (const float*)d_in[4];
    const float* blk_w  = (const float*)d_in[5];
    const float* blk_lw = (const float*)d_in[6];
    const float* blk_b  = (const float*)d_in[7];
    const float* w_out  = (const float*)d_in[8];
    const float* lw_out = (const float*)d_in[9];
    const float* b_out  = (const float*)d_in[10];

    char* ws = (char*)d_ws;
    unsigned short* adjb = (unsigned short*)ws;  ws += (size_t)N * N * 2;       // 128 MB
    unsigned short* ytT  = (unsigned short*)ws;  ws += (size_t)B * H * N * 2;   //  16 MB
    float* xlwT          = (float*)ws;           ws += (size_t)B * H * N * 4;   //  32 MB
    float* xA            = (float*)ws;           ws += (size_t)B * N * H * 4;   //  32 MB
    float* xB            = (float*)ws;                                          //  32 MB

    float* x_out   = (float*)d_out;                       // [B,N,3]
    float* x_final = (float*)d_out + (size_t)B * N * OUT; // [B,N,64]

    {   // adj -> bf16 once
        long n = (long)N * N;
        cvt_adj_bf16<<<(unsigned)(n / (4 * 256)), 256, 0, stream>>>(adj, adjb, n);
    }

    dim3 gProj(N / 64, B), gGemm(N / 256, B);

    auto layer = [&](const float* xin, int Din, const float* w, const float* lw,
                     const float* bias, float* out, int outDim, bool relu,
                     const float* resid) {
        gcn_proj<<<gProj, 256, 0, stream>>>(xin, w, lw, ytT, xlwT, Din, outDim);
        if (outDim == H)
            gcn_wmma<4><<<gGemm, 256, 0, stream>>>(adjb, ytT, xlwT, bias, resid,
                                                   out, outDim, relu ? 1 : 0,
                                                   resid ? 1 : 0);
        else
            gcn_wmma<1><<<gGemm, 256, 0, stream>>>(adjb, ytT, xlwT, bias, resid,
                                                   out, outDim, relu ? 1 : 0,
                                                   resid ? 1 : 0);
    };

    // input layer
    layer(verts, IN, w_in, lw_in, b_in, xA, H, true, nullptr);

    // residual blocks: h = relu(gcn(x)); h = relu(gcn(h)); x = (x + h) * 0.5
    for (int i = 0; i < NB; ++i) {
        const float* w0  = blk_w  + ((size_t)i * 2 + 0) * H * H;
        const float* lw0 = blk_lw + ((size_t)i * 2 + 0) * H * H;
        const float* b0  = blk_b  + ((size_t)i * 2 + 0) * H;
        const float* w1  = blk_w  + ((size_t)i * 2 + 1) * H * H;
        const float* lw1 = blk_lw + ((size_t)i * 2 + 1) * H * H;
        const float* b1  = blk_b  + ((size_t)i * 2 + 1) * H;
        layer(xA, H, w0, lw0, b0, xB, H, true, nullptr);
        layer(xB, H, w1, lw1, b1, xA, H, true, xA);  // elementwise-safe in-place
    }

    // output layer (no relu, Dout=3 -> one 16-col tile, masked stores)
    layer(xA, H, w_out, lw_out, b_out, x_out, OUT, false, nullptr);

    {   // second tuple element: final hidden state x
        long n = (long)B * N * H;
        copy_f32<<<(unsigned)((n + 255) / 256), 256, 0, stream>>>(xA, x_final, n);
    }
}
